// ModelClass_9526237463078
// MI455X (gfx1250) — matrix operand order repaired and checked by reading, not compiled, not run on hardware
//
#include <hip/hip_runtime.h>
#include <math.h>

// ---------------------------------------------------------------------------
// MI455X (gfx1250). Heavy part = per-node FFN GEMMs (M up to 131072 rows,
// K/N in {3..512}) lowered to v_wmma_f32_16x16x32_bf16 with f32 accumulate.
// Wave tile 32x32 (4 WMMA per 32-deep K step). Weights staged transposed in
// LDS as bf16 (vector ds loads/stores); A fragments via global_load_b128 on
// the K%32==0 fast path; index-clamped (branch-free) loads on ragged shapes.
// ---------------------------------------------------------------------------

typedef __attribute__((ext_vector_type(16))) __bf16 v16bf;
typedef __attribute__((ext_vector_type(8)))  float  v8f;

__device__ __forceinline__ __bf16 f2bf(float f) {
  unsigned u = __builtin_bit_cast(unsigned, f);
  u += 0x7FFFu + ((u >> 16) & 1u);          // round-to-nearest-even
  unsigned short s = (unsigned short)(u >> 16);
  return __builtin_bit_cast(__bf16, s);
}
__device__ __forceinline__ unsigned short f2bfu(float f) {
  unsigned u = __builtin_bit_cast(unsigned, f);
  u += 0x7FFFu + ((u >> 16) & 1u);
  return (unsigned short)(u >> 16);
}

// Y[M,N] = act(X[M,K] @ W[K,N] + bias[N]); leakyrelu(slope) if doAct.
// Block: 256 threads = 8 waves. Block tile: 256 rows x 32 cols.
// Wave w owns rows [blockIdx.x*256 + 32w, +32) as two 16-row A fragments and
// both 16-col B fragments -> 4 accumulators, 4 v_wmma per K chunk.
__global__ __launch_bounds__(256) void k_gemm_wmma(
    const float* __restrict__ X, const float* __restrict__ W,
    const float* __restrict__ bias, float* __restrict__ Y,
    int M, int K, int N, float slope, int doAct)
{
  __shared__ __attribute__((aligned(32))) __bf16 sWT[32 * 32]; // [n][k] chunk
  const int tid  = threadIdx.x;
  const int wave = tid >> 5;
  const int lane = tid & 31;
  const int lm   = lane & 15;
  const int hi   = lane >> 4;
  const int col0 = blockIdx.y * 32;
  const int row0 = blockIdx.x * 256 + wave * 32;
  const bool fastK = (K & 31) == 0;

  v8f c00 = {}, c01 = {}, c10 = {}, c11 = {};

  // Row pointers clamped into range: rows >= M compute garbage that is never
  // stored (epilogue guards), so no per-element guard is needed.
  const int m0 = row0 + lm;
  const int m1 = row0 + 16 + lm;
  const float* xrow0 = X + (size_t)(m0 < M ? m0 : 0) * K;
  const float* xrow1 = X + (size_t)(m1 < M ? m1 : 0) * K;

  // LDS stage mapping: thread t covers 4 consecutive k for one n.
  const int sn  = tid >> 3;          // 0..31
  const int sk  = (tid & 7) * 4;     // 0,4,...,28
  const int nn  = col0 + sn;
  const int nnc = (nn < N) ? nn : (N - 1);
  const bool nok = nn < N;

  for (int k0 = 0; k0 < K; k0 += 32) {
    // ---- stage W[k0:k0+32, col0:col0+32] transposed into LDS (bf16) ----
    // 32-bit index math: W tiles are at most 512x512 elements.
    {
      unsigned short h[4];
      if (fastK) {
        int wi = (k0 + sk) * N + nnc;            // k always in range
#pragma unroll
        for (int i = 0; i < 4; ++i) {
          float w = W[wi]; wi += N;
          h[i] = f2bfu(nok ? w : 0.0f);
        }
      } else {
#pragma unroll
        for (int i = 0; i < 4; ++i) {
          int kk  = k0 + sk + i;
          int kkc = (kk < K) ? kk : (K - 1);
          float w = W[kkc * N + nnc];            // unconditional load
          w = (kk < K && nok) ? w : 0.0f;        // zero pad (v_cndmask)
          h[i] = f2bfu(w);
        }
      }
      uint2 pk;
      pk.x = (unsigned)h[0] | ((unsigned)h[1] << 16);
      pk.y = (unsigned)h[2] | ((unsigned)h[3] << 16);
      *(uint2*)(&sWT[sn * 32 + sk]) = pk;        // ds_store_b64
    }
    __syncthreads();

    // ---- A fragments (16-bit A 16x32 layout) ----
    // lanes 0-15: K = k0+[0..15]; lanes 16-31: K = k0+[16..31]
    v16bf a0, a1;
    const int kb = k0 + hi * 16;
    if (fastK) {
      const float* p0 = xrow0 + kb;
      const float* p1 = xrow1 + kb;
      if (k0 + 32 < K) { __builtin_prefetch(p0 + 32, 0, 1); }
      float4 q0 = *(const float4*)(p0);
      float4 q1 = *(const float4*)(p0 + 4);
      float4 q2 = *(const float4*)(p0 + 8);
      float4 q3 = *(const float4*)(p0 + 12);
      a0[0] = f2bf(q0.x); a0[1] = f2bf(q0.y); a0[2]  = f2bf(q0.z); a0[3]  = f2bf(q0.w);
      a0[4] = f2bf(q1.x); a0[5] = f2bf(q1.y); a0[6]  = f2bf(q1.z); a0[7]  = f2bf(q1.w);
      a0[8] = f2bf(q2.x); a0[9] = f2bf(q2.y); a0[10] = f2bf(q2.z); a0[11] = f2bf(q2.w);
      a0[12] = f2bf(q3.x); a0[13] = f2bf(q3.y); a0[14] = f2bf(q3.z); a0[15] = f2bf(q3.w);
      q0 = *(const float4*)(p1);
      q1 = *(const float4*)(p1 + 4);
      q2 = *(const float4*)(p1 + 8);
      q3 = *(const float4*)(p1 + 12);
      a1[0] = f2bf(q0.x); a1[1] = f2bf(q0.y); a1[2]  = f2bf(q0.z); a1[3]  = f2bf(q0.w);
      a1[4] = f2bf(q1.x); a1[5] = f2bf(q1.y); a1[6]  = f2bf(q1.z); a1[7]  = f2bf(q1.w);
      a1[8] = f2bf(q2.x); a1[9] = f2bf(q2.y); a1[10] = f2bf(q2.z); a1[11] = f2bf(q2.w);
      a1[12] = f2bf(q3.x); a1[13] = f2bf(q3.y); a1[14] = f2bf(q3.z); a1[15] = f2bf(q3.w);
    } else {
      // Ragged K (K=3,6): clamp indices; B is zero-padded so values at k>=K
      // cannot contribute. Branch-free.
#pragma unroll
      for (int j = 0; j < 8; ++j) {
        int k1 = kb + j, k2 = kb + 8 + j;
        int c1 = (k1 < K) ? k1 : 0;
        int c2 = (k2 < K) ? k2 : 0;
        a0[j]     = f2bf(xrow0[c1]);
        a0[8 + j] = f2bf(xrow0[c2]);
        a1[j]     = f2bf(xrow1[c1]);
        a1[8 + j] = f2bf(xrow1[c2]);
      }
    }

    // ---- B fragments: contiguous 32B LDS reads (transposed layout) ----
    v16bf b0 = *(const v16bf*)(&sWT[lm * 32 + hi * 16]);
    v16bf b1 = *(const v16bf*)(&sWT[(16 + lm) * 32 + hi * 16]);

    c00 = __builtin_amdgcn_wmma_f32_16x16x32_bf16(false, a0, false, b0, (short)0, c00, false, false);
    c01 = __builtin_amdgcn_wmma_f32_16x16x32_bf16(false, a0, false, b1, (short)0, c01, false, false);
    c10 = __builtin_amdgcn_wmma_f32_16x16x32_bf16(false, a1, false, b0, (short)0, c10, false, false);
    c11 = __builtin_amdgcn_wmma_f32_16x16x32_bf16(false, a1, false, b1, (short)0, c11, false, false);
    __syncthreads();
  }

  // ---- epilogue: C layout lanes 0-15 rows 0..7, lanes 16-31 rows 8..15 ----
  const int nA = col0 + lm;
  const int nB = col0 + 16 + lm;
  const float bvA = (nA < N && bias) ? bias[nA] : 0.0f;
  const float bvB = (nB < N && bias) ? bias[nB] : 0.0f;
#pragma unroll
  for (int r = 0; r < 8; ++r) {
    int mA = row0 + hi * 8 + r;        // tile row for a0
    int mB = row0 + 16 + hi * 8 + r;   // tile row for a1
    if (mA < M) {
      if (nA < N) { float v = c00[r] + bvA; if (doAct) v = v > 0.f ? v : v * slope; Y[(size_t)mA * N + nA] = v; }
      if (nB < N) { float v = c01[r] + bvB; if (doAct) v = v > 0.f ? v : v * slope; Y[(size_t)mA * N + nB] = v; }
    }
    if (mB < M) {
      if (nA < N) { float v = c10[r] + bvA; if (doAct) v = v > 0.f ? v : v * slope; Y[(size_t)mB * N + nA] = v; }
      if (nB < N) { float v = c11[r] + bvB; if (doAct) v = v > 0.f ? v : v * slope; Y[(size_t)mB * N + nB] = v; }
    }
  }
}

// out[b, 0..F) = max over nodes, out[b, F..2F) = sum over nodes.
__global__ void k_pool_maxsum(const float* __restrict__ x, float* __restrict__ out,
                              int Nn, int F) {
  int b = blockIdx.x;
  for (int f = threadIdx.x; f < F; f += blockDim.x) {
    float mx = -INFINITY, sm = 0.f;
    const float* p = x + (size_t)b * Nn * F + f;
    for (int n = 0; n < Nn; ++n) { float v = p[(size_t)n * F]; mx = fmaxf(mx, v); sm += v; }
    out[(size_t)b * 2 * F + f]     = mx;
    out[(size_t)b * 2 * F + F + f] = sm;
  }
}

// lat[b, colOff + (sum | max | std)] per channel; std = population std.
__global__ void k_latpool(const float* __restrict__ x, float* __restrict__ lat,
                          int Nn, int F, int latLD, int colOff) {
  int b = blockIdx.x;
  for (int f = threadIdx.x; f < F; f += blockDim.x) {
    float mx = -INFINITY, sm = 0.f, sq = 0.f;
    const float* p = x + (size_t)b * Nn * F + f;
    for (int n = 0; n < Nn; ++n) { float v = p[(size_t)n * F]; mx = fmaxf(mx, v); sm += v; sq += v * v; }
    float mean = sm / (float)Nn;
    float var  = fmaxf(sq / (float)Nn - mean * mean, 0.f);
    float* L = lat + (size_t)b * latLD + colOff;
    L[f] = sm; L[F + f] = mx; L[2 * F + f] = sqrtf(var);
  }
}

// z[b,n,:] = [h[b,n,0..L), g[b,0..G)]
__global__ void k_concat_hg(const float* __restrict__ h, const float* __restrict__ g,
                            float* __restrict__ z, int Nn, int L, int G, size_t total) {
  size_t i = (size_t)blockIdx.x * blockDim.x + threadIdx.x;
  if (i >= total) return;
  int ZW = L + G;
  size_t row = i / ZW;
  int c = (int)(i % ZW);
  int b = (int)(row / Nn);
  z[i] = (c < L) ? h[row * (size_t)L + c] : g[(size_t)b * G + (c - L)];
}

__global__ void k_add(float* __restrict__ y, const float* __restrict__ a, size_t n) {
  size_t i = (size_t)blockIdx.x * blockDim.x + threadIdx.x;
  if (i < n) y[i] += a[i];
}
__global__ void k_copy(float* __restrict__ y, const float* __restrict__ a, size_t n) {
  size_t i = (size_t)blockIdx.x * blockDim.x + threadIdx.x;
  if (i < n) y[i] = a[i];
}
__global__ void k_zero(float* __restrict__ y, size_t n) {
  size_t i = (size_t)blockIdx.x * blockDim.x + threadIdx.x;
  if (i < n) y[i] = 0.0f;
}

// crit[b] += o[b,0]; cond[b,:] += o[b,1:6] / 3  (mean over 3 levels)
__global__ void k_disc_accum(const float* __restrict__ o, float* __restrict__ crit,
                             float* __restrict__ cond, int B) {
  int b = blockIdx.x * blockDim.x + threadIdx.x;
  if (b < B) {
    crit[b] += o[b * 6];
    for (int j = 0; j < 5; ++j) cond[b * 5 + j] += o[b * 6 + 1 + j] * (1.0f / 3.0f);
  }
}

// GATv2 BipartPool attention. xl:[B,Nn,H,C], xr:[R,H,C], att:[H,C] -> out:[B,R,C]
__global__ __launch_bounds__(256) void k_bipart(
    const float* __restrict__ xl, const float* __restrict__ xr,
    const float* __restrict__ att, const float* __restrict__ bias,
    float* __restrict__ out, int Nn, int C, int R)
{
  const int H = 2;
  extern __shared__ float smem[];
  float* se   = smem;        // [Nn]  scores -> exp weights
  float* sacc = smem + Nn;   // [H*C] per-head accumulators
  __shared__ float sred[8];
  int b = blockIdx.x, r = blockIdx.y, tid = threadIdx.x;
  const int HC = H * C;
  const float* xrr = xr + (size_t)r * HC;

  float lmax = -INFINITY;
  for (int n = tid; n < Nn; n += blockDim.x) {
    const float* xln = xl + ((size_t)b * Nn + n) * HC;
    float e = 0.f;
    for (int i = 0; i < HC; ++i) {
      float zv = xln[i] + xrr[i];
      zv = (zv > 0.f) ? zv : zv * 0.2f;
      e += att[i] * zv;
    }
    se[n] = e;
    lmax = fmaxf(lmax, e);
  }
  for (int off = 16; off; off >>= 1) lmax = fmaxf(lmax, __shfl_xor(lmax, off, 32));
  if ((tid & 31) == 0) sred[tid >> 5] = lmax;
  __syncthreads();
  float bmax = sred[0];
  for (int w = 1; w < (int)(blockDim.x >> 5); ++w) bmax = fmaxf(bmax, sred[w]);
  __syncthreads();

  float lsum = 0.f;
  for (int n = tid; n < Nn; n += blockDim.x) {
    float w = __expf(se[n] - bmax);
    se[n] = w;
    lsum += w;
  }
  for (int off = 16; off; off >>= 1) lsum += __shfl_xor(lsum, off, 32);
  if ((tid & 31) == 0) sred[tid >> 5] = lsum;
  __syncthreads();
  float bsum = 0.f;
  for (int w = 0; w < (int)(blockDim.x >> 5); ++w) bsum += sred[w];
  __syncthreads();

  for (int i = tid; i < HC; i += blockDim.x) {
    float acc = 0.f;
    const float* xp = xl + (size_t)b * Nn * HC + i;
    for (int n = 0; n < Nn; ++n) acc += se[n] * xp[(size_t)n * HC];
    sacc[i] = acc / bsum;
  }
  __syncthreads();
  for (int cc = tid; cc < C; cc += blockDim.x)
    out[((size_t)b * R + r) * C + cc] = 0.5f * (sacc[cc] + sacc[C + cc]) + bias[cc];
}

// ---------------------------------------------------------------------------

extern "C" void kernel_launch(void* const* d_in, const int* in_sizes, int n_in,
                              void* d_out, int out_size, void* d_ws, size_t ws_size,
                              hipStream_t stream) {
  (void)in_sizes; (void)n_in; (void)out_size; (void)ws_size;
  const int B = 128;

  auto P = [&](int i) { return (const float*)d_in[i]; };

  // ---- parameter index map: jax tree_leaves order (sorted dict keys) ----
  struct FFNIdx { int w0, b0, w1, b1, w2, b2; };
  struct CNUIdx { FFNIdx emb, glob, out; };
  auto FFN = [](int base) { return FFNIdx{base, base + 1, base + 2, base + 3, base + 4, base + 5}; };
  auto CNU = [&](int base) { return CNUIdx{FFN(base), FFN(base + 6), FFN(base + 12)}; };

  CNUIdx emb0_cnu = CNU(2);   FFNIdx emb0_inp = FFN(20);
  CNUIdx emb1_cnu = CNU(26);  FFNIdx emb1_inp = FFN(44);
  CNUIdx pd0_c0 = CNU(50),  pd0_c1 = CNU(68);  FFNIdx pd0_disc = FFN(86);
  CNUIdx pd1_c0 = CNU(92),  pd1_c1 = CNU(110); FFNIdx pd1_disc = FFN(128);
  CNUIdx pd2_c0 = CNU(134), pd2_c1 = CNU(152); FFNIdx pd2_disc = FFN(170);
  const int p0_att = 176, p0_bias = 177, p0_bl = 178, p0_br = 179, p0_wl = 180, p0_wr = 181, p0_xc = 182;
  const int p1_att = 183, p1_bias = 184, p1_bl = 185, p1_br = 186, p1_wl = 187, p1_wr = 188, p1_xc = 189;

  // ---- outputs: crit [B,1] ++ lat [B,1161] ++ cond [B,5] ----
  float* crit = (float*)d_out;
  float* lat  = crit + B;
  float* cond = lat + (size_t)B * 1161;

  // ---- workspace ----
  const size_t M0 = (size_t)B * 1024, M1 = (size_t)B * 32, M2 = (size_t)B * 8;
  const size_t REG = M0 * 128;
  float* r0 = (float*)d_ws;
  float* r1 = r0 + REG;   // r1..r2 reused as level-0 xl [M0,256]
  float* r2 = r1 + REG;
  float* r3 = r2 + REG;
  float* r4 = r3 + REG;
  float* r5 = r4 + REG;
  float* sm0 = r5 + REG;
  float* s_x1 = sm0;
  float* s_x2 = s_x1 + M1 * 128;
  float* s_t  = s_x2 + M2 * 256;
  float* s_a  = s_t  + M1 * 256;
  float* s_b  = s_a  + M1 * 256;
  float* s_h  = s_b  + M1 * 256;
  float* s_z  = s_h  + M1 * 256;
  float* s_o  = s_z  + M1 * 256;
  float* s_xl = s_o  + M1 * 256;
  float* s_xr = s_xl + M1 * 512;
  float* s_ag = s_xr + 32 * 512;
  float* s_g1 = s_ag + (size_t)B * 512;
  float* s_g2 = s_g1 + (size_t)B * 128;
  float* s_g  = s_g2 + (size_t)B * 128;
  float* s_d  = s_g  + (size_t)B * 128;

  // ---- launch helpers ----
  auto gemm = [&](const float* X, int wi, int bi, float* Y, int M, int K, int N, bool act) {
    dim3 grid((unsigned)((M + 255) / 256), (unsigned)((N + 31) / 32));
    k_gemm_wmma<<<grid, dim3(256), 0, stream>>>(X, P(wi), P(bi), Y, M, K, N, 0.01f, act ? 1 : 0);
  };
  auto ffn = [&](const float* X, FFNIdx f, float* t1, float* t2, float* Y,
                 int M, int K, int Nout, bool finalLinear) {
    gemm(X,  f.w0, f.b0, t1, M, K,   128, true);
    gemm(t1, f.w1, f.b1, t2, M, 128, 128, true);
    gemm(t2, f.w2, f.b2, Y,  M, 128, Nout, !finalLinear);
  };
  auto zero = [&](float* p, size_t n) { k_zero<<<(unsigned)((n + 255) / 256), 256, 0, stream>>>(p, n); };
  auto copyb = [&](float* d, const float* s, size_t n) { k_copy<<<(unsigned)((n + 255) / 256), 256, 0, stream>>>(d, s, n); };
  auto addb = [&](float* d, const float* s, size_t n) { k_add<<<(unsigned)((n + 255) / 256), 256, 0, stream>>>(d, s, n); };
  auto cnu = [&](const float* X, CNUIdx ci, float* t1, float* t2, float* hb,
                 float* zb, float* Ob, int Nn, int F) {
    int M = B * Nn;
    ffn(X, ci.emb, t1, t2, hb, M, F, 64, false);
    k_pool_maxsum<<<B, 256, 0, stream>>>(hb, s_ag, Nn, 64);
    ffn(s_ag, ci.glob, s_g1, s_g2, s_g, B, 128, 64, false);
    size_t tot = (size_t)M * 128;
    k_concat_hg<<<(unsigned)((tot + 255) / 256), 256, 0, stream>>>(hb, s_g, zb, Nn, 64, 64, tot);
    ffn(zb, ci.out, t1, t2, Ob, M, 128, F, true);
  };
  auto tsumdisc = [&](float* t, CNUIdx c0, CNUIdx c1, FFNIdx disc,
                      float* t1, float* t2, float* hb, float* zb, float* Ob,
                      int Nn, int F) {
    int M = B * Nn;
    cnu(t, c0, t1, t2, hb, zb, Ob, Nn, F); addb(t, Ob, (size_t)M * F);
    cnu(t, c1, t1, t2, hb, zb, Ob, Nn, F); addb(t, Ob, (size_t)M * F);
    k_pool_maxsum<<<B, 256, 0, stream>>>(t, s_ag, Nn, F);
    ffn(s_ag, disc, s_g1, s_g2, s_d, B, 2 * F, 6, true);
    k_disc_accum<<<1, 128, 0, stream>>>(s_d, crit, cond, B);
  };

  const float* x0 = P(0);
  zero(crit, (size_t)B);
  zero(cond, (size_t)B * 5);

  // ==================== level 0 (N=1024, F=3) ====================
  k_latpool<<<B, 256, 0, stream>>>(x0, lat, 1024, 3, 1161, 0);
  copyb(r0, x0, M0 * 3);
  tsumdisc(r0, pd0_c0, pd0_c1, pd0_disc, r1, r2, r3, r4, r5, 1024, 3);
  ffn(x0, emb0_inp, r1, r2, r3, (int)M0, 3, 128, true);           // xp in r3
  cnu(r3, emb0_cnu, r1, r2, r4, r5, r0, 1024, 128);               // cnu out in r0
  addb(r0, r3, M0 * 128);                                          // x1pre in r0
  gemm(r0, p0_wl, p0_bl, r1, (int)M0, 128, 256, false);           // xl spans r1..r2
  gemm(P(p0_xc), p0_wr, p0_br, s_xr, 32, 128, 256, false);
  k_bipart<<<dim3(B, 32), 256, (1024 + 256) * sizeof(float), stream>>>(
      r1, s_xr, P(p0_att), P(p0_bias), s_x1, 1024, 128, 32);

  // ==================== level 1 (N=32, F=128) ====================
  k_latpool<<<B, 256, 0, stream>>>(s_x1, lat, 32, 128, 1161, 9);
  copyb(s_t, s_x1, M1 * 128);
  tsumdisc(s_t, pd1_c0, pd1_c1, pd1_disc, s_a, s_b, s_h, s_z, s_o, 32, 128);
  ffn(s_x1, emb1_inp, s_a, s_b, s_z, (int)M1, 128, 256, true);    // xp in s_z
  cnu(s_z, emb1_cnu, s_a, s_b, s_h, s_t, s_o, 32, 256);
  addb(s_o, s_z, M1 * 256);                                        // x2pre in s_o
  gemm(s_o, p1_wl, p1_bl, s_xl, (int)M1, 256, 512, false);
  gemm(P(p1_xc), p1_wr, p1_br, s_xr, 8, 256, 512, false);
  k_bipart<<<dim3(B, 8), 256, (32 + 512) * sizeof(float), stream>>>(
      s_xl, s_xr, P(p1_att), P(p1_bias), s_x2, 32, 256, 8);

  // ==================== level 2 (N=8, F=256) ====================
  k_latpool<<<B, 256, 0, stream>>>(s_x2, lat, 8, 256, 1161, 393);
  copyb(s_t, s_x2, M2 * 256);
  tsumdisc(s_t, pd2_c0, pd2_c1, pd2_disc, s_a, s_b, s_h, s_z, s_o, 8, 256);
}
